// RandomProjectionQuantizer_82136954568864
// MI455X (gfx1250) — matrix operand order repaired and checked by reading, not compiled, not run on hardware
//
#include <hip/hip_runtime.h>

#define Bb 8
#define Ll 2048
#define Dd 1024
#define Qq 256
#define Cc 4096
#define Mm (Bb * Ll)

#define TP 264        // LDS pitch (halves) for the T tile (bank-conflict padding)
#define CP 264        // LDS pitch (halves) for the codebook chunk (256 data + 8 pad)
#define CHUNK 64      // columns of C per phase-2 chunk
#define CBT_PITCH 256 // halves per row of transposed codebook in global ws

typedef _Float16 v16h __attribute__((ext_vector_type(16)));
typedef _Float16 h8   __attribute__((ext_vector_type(8)));
typedef float    v8f  __attribute__((ext_vector_type(8)));
typedef int      i4v  __attribute__((vector_size(16))); // matches builtin's b128 pointee

#if defined(__has_builtin)
#if __has_builtin(__builtin_amdgcn_global_load_async_to_lds_b128)
#define USE_ASYNC_LDS 1
#endif
#endif

__device__ __forceinline__ v8f wmma_f16(v16h a, v16h b, v8f c) {
  return __builtin_amdgcn_wmma_f32_16x16x32_f16(
      /*neg_a=*/false, a, /*neg_b=*/false, b,
      /*c_mod=*/(short)0, c, /*reuse_a=*/false, /*reuse_b=*/false);
}

__device__ __forceinline__ void cvt8(v16h& v, int base, float4 a, float4 b) {
  v[base + 0] = (_Float16)a.x; v[base + 1] = (_Float16)a.y;
  v[base + 2] = (_Float16)a.z; v[base + 3] = (_Float16)a.w;
  v[base + 4] = (_Float16)b.x; v[base + 5] = (_Float16)b.y;
  v[base + 6] = (_Float16)b.z; v[base + 7] = (_Float16)b.w;
}

// A fragment (16x32 f16) from fp32 global: lane holds K = base+khalf*16+{0..15}
__device__ __forceinline__ v16h loadA_f32(const float* p) {
  float4 a0 = *(const float4*)(p);
  float4 a1 = *(const float4*)(p + 4);
  float4 a2 = *(const float4*)(p + 8);
  float4 a3 = *(const float4*)(p + 12);
  v16h A;
  cvt8(A, 0, a0, a1);
  cvt8(A, 8, a2, a3);
  return A;
}

// 16 contiguous halves -> v16h (B fragment lane data, or A from f16 memory)
__device__ __forceinline__ v16h load16h(const _Float16* p) {
  h8 lo = *(const h8*)(p);
  h8 hi = *(const h8*)(p + 8);
  return __builtin_shufflevector(lo, hi, 0,1,2,3,4,5,6,7,8,9,10,11,12,13,14,15);
}

// 16-byte global -> LDS copy: async on CDNA5 (ASYNCcnt), sync fallback otherwise
__device__ __forceinline__ void copy16_g2l(const void* gsrc, void* ldst) {
#ifdef USE_ASYNC_LDS
  __builtin_amdgcn_global_load_async_to_lds_b128(
      (__attribute__((address_space(1))) i4v*)gsrc,
      (__attribute__((address_space(3))) i4v*)ldst, 0, 0);
#else
  *(h8*)ldst = *(const h8*)gsrc;
#endif
}

__device__ __forceinline__ void wait_async_copies() {
#ifdef USE_ASYNC_LDS
#if __has_builtin(__builtin_amdgcn_s_wait_asynccnt)
  __builtin_amdgcn_s_wait_asynccnt(0);
#else
  asm volatile("s_wait_asynccnt 0" ::: "memory");
#endif
#endif
}

// ---------------- prep kernels (one-time, tiny vs main compute) ----------------

// cb_sq[c] = sum_q codebook[q][c]^2   (coalesced)
__global__ void rpq_cbsq_kernel(const float* __restrict__ cb, float* __restrict__ cbsq) {
  int c = blockIdx.x * blockDim.x + threadIdx.x;
  float s = 0.0f;
#pragma unroll 8
  for (int q = 0; q < Qq; ++q) {
    float v = cb[(size_t)q * Cc + c];
    s += v * v;
  }
  cbsq[c] = s;
}

// cbTg[c][k] = (f16) codebook[k][c]  -- transposed f16 codebook, pitch 256 halves
__global__ void rpq_prep_cb(const float* __restrict__ cb, _Float16* __restrict__ cbTg) {
  int flat = blockIdx.x * blockDim.x + threadIdx.x; // Qq*Cc threads
  int c = flat & (Cc - 1);
  int k = flat >> 12; // /Cc
  cbTg[(size_t)c * CBT_PITCH + k] = (_Float16)cb[(size_t)k * Cc + c];
}

// Wh = (f16) W, same (Q,D) row-major layout
__global__ void rpq_prep_w(const float* __restrict__ W, _Float16* __restrict__ Wh) {
  int i = blockIdx.x * blockDim.x + threadIdx.x;
  Wh[i] = (_Float16)W[i];
}

// ---------------- fused main kernel ----------------

__global__ void __launch_bounds__(256)
rpq_main_kernel(const float* __restrict__ X, const _Float16* __restrict__ Wh,
                const _Float16* __restrict__ cbTg, const float* __restrict__ cbsq,
                float* __restrict__ out) {
  // One region, time-multiplexed: T tile (16896 B) -> codebook chunk (33792 B) -> argmin (16 KB)
  __shared__ __align__(16) _Float16 smem[CHUNK * CP];

  const int tid   = threadIdx.x;
  const int w     = tid >> 5; // wave id, 0..7 (wave32)
  const int l     = tid & 31;
  const int nlo   = l & 15;
  const int khalf = l >> 4;
  const int m0    = blockIdx.x * 32;

  // ---- Phase 1: T = X_tile(32x1024) * Wh^T -> 32x256, each wave 2m x 2n tiles ----
  {
    v8f acc00 = {0,0,0,0,0,0,0,0}, acc01 = {0,0,0,0,0,0,0,0};
    v8f acc10 = {0,0,0,0,0,0,0,0}, acc11 = {0,0,0,0,0,0,0,0};
    const float*    x0 = X  + (size_t)(m0 + nlo) * Dd;       // m-subtile 0 row
    const float*    x1 = x0 + (size_t)16 * Dd;               // m-subtile 1 row
    const _Float16* w0 = Wh + (size_t)(w * 32 + nlo) * Dd;   // n-tile 0 (q row)
    const _Float16* w1 = w0 + (size_t)16 * Dd;               // n-tile 1

#pragma unroll 1
    for (int kk = 0; kk < Dd / 32; ++kk) {
      const int kb = kk * 32;
      v16h A0 = loadA_f32(x0 + kb + khalf * 16);
      v16h A1 = loadA_f32(x1 + kb + khalf * 16);
      v16h B0 = load16h(w0 + kb + khalf * 16);
      v16h B1 = load16h(w1 + kb + khalf * 16);
      acc00 = wmma_f16(A0, B0, acc00);
      acc01 = wmma_f16(A0, B1, acc01);
      acc10 = wmma_f16(A1, B0, acc10);
      acc11 = wmma_f16(A1, B1, acc11);
    }
    // Store T to LDS as f16 (C-layout: VGPR r, lane -> row r+8*khalf, col nlo)
    _Float16* Tl = smem;
#pragma unroll
    for (int r = 0; r < 8; ++r) {
      int row0 = r + 8 * khalf;
      int col0 = w * 32 + nlo;
      Tl[(row0)      * TP + col0]      = (_Float16)acc00[r];
      Tl[(row0)      * TP + col0 + 16] = (_Float16)acc01[r];
      Tl[(row0 + 16) * TP + col0]      = (_Float16)acc10[r];
      Tl[(row0 + 16) * TP + col0 + 16] = (_Float16)acc11[r];
    }
  }
  __syncthreads();

  // ---- Load this wave's A fragments (16x256 of T) into registers ----
  const int msub2 = w >> 2; // 16-row subtile of T
  const int cquad = w & 3;  // 16-col tile within 64-col chunk
  v16h Af[8];
  {
    const _Float16* trow = smem + (size_t)(msub2 * 16 + nlo) * TP + khalf * 16;
#pragma unroll
    for (int ks = 0; ks < 8; ++ks) {
      h8 lo = *(const h8*)(trow + ks * 32);
      h8 hi = *(const h8*)(trow + ks * 32 + 8);
      Af[ks] = __builtin_shufflevector(lo, hi, 0,1,2,3,4,5,6,7,8,9,10,11,12,13,14,15);
    }
  }
  __syncthreads(); // T consumed; smem now becomes the codebook chunk buffer

  // ---- Phase 2: scores over C in 64-col chunks, running argmin ----
  float bestv[8];
  int   besti[8];
#pragma unroll
  for (int r = 0; r < 8; ++r) { bestv[r] = 3.4e38f; besti[r] = 0; }

  const char* gbase = (const char*)cbTg;
  char*       lbase = (char*)smem;
  const int NCHUNK = Cc / CHUNK;

  for (int chunk = 0; chunk < NCHUNK; ++chunk) {
    const int c0 = chunk * CHUNK;

    // Copy 64 rows x 512 B of cbTg into LDS (padded pitch). Wave w: rows w*8..w*8+7;
    // each lane moves 16 B -> one global_load_async_to_lds_b128 per row per wave.
#pragma unroll
    for (int j = 0; j < 8; ++j) {
      int row = w * 8 + j;
      copy16_g2l(gbase + ((size_t)(c0 + row) * (CBT_PITCH * 2)) + l * 16,
                 lbase + ((size_t)row * (CP * 2)) + l * 16);
    }
    // prefetch next chunk's rows (threads 0..63), L2-resident but cheap insurance
    if (chunk + 1 < NCHUNK && tid < CHUNK)
      __builtin_prefetch(gbase + (size_t)(c0 + CHUNK + tid) * (CBT_PITCH * 2), 0, 0);
    wait_async_copies();
    __syncthreads();

    v8f acc = {0,0,0,0,0,0,0,0};
    const _Float16* bbase = smem + (size_t)(cquad * 16 + nlo) * CP + khalf * 16;
#pragma unroll
    for (int ks = 0; ks < 8; ++ks) {
      v16h Bf = load16h(bbase + ks * 32);
      acc = wmma_f16(Af[ks], Bf, acc);
    }

    const int cglob = c0 + cquad * 16 + nlo;
    const float csq = cbsq[cglob];
#pragma unroll
    for (int r = 0; r < 8; ++r) {
      float s = csq - 2.0f * acc[r];
      if (s < bestv[r]) { bestv[r] = s; besti[r] = cglob; } // strict <: earliest c wins
    }
    __syncthreads(); // all waves done reading chunk before next overwrite
  }

  // ---- Argmin reduction across lanes/waves (reuse smem region) ----
  float* redv = (float*)smem;            // 32*64 floats = 8 KB
  int*   redi = (int*)(redv + 32 * 64);  // 8 KB; total 16 KB <= 33792 B
#pragma unroll
  for (int r = 0; r < 8; ++r) {
    int row  = msub2 * 16 + r + 8 * khalf;
    int slot = cquad * 16 + nlo;
    redv[row * 64 + slot] = bestv[r];
    redi[row * 64 + slot] = besti[r];
  }
  __syncthreads();
  if (tid < 32) {
    float bv = 3.4e38f;
    int   bi = 0x7fffffff;
    for (int j = 0; j < 64; ++j) {
      float v = redv[tid * 64 + j];
      int   i = redi[tid * 64 + j];
      if (v < bv || (v == bv && i < bi)) { bv = v; bi = i; }
    }
    out[m0 + tid] = (float)bi;
  }
}

extern "C" void kernel_launch(void* const* d_in, const int* in_sizes, int n_in,
                              void* d_out, int out_size, void* d_ws, size_t ws_size,
                              hipStream_t stream) {
  const float* X  = (const float*)d_in[0]; // (B,L,D) fp32
  // d_in[1]: mask_time_indices (unused by label computation)
  const float* W  = (const float*)d_in[2]; // (Q,D) fp32
  const float* cb = (const float*)d_in[3]; // (Q,C) fp32

  // workspace carve-up: [cbsq: 16 KB][cbTg: 2 MB][Wh: 512 KB]  (~2.6 MB total)
  float*    cbsq = (float*)d_ws;
  _Float16* cbTg = (_Float16*)((char*)d_ws + (size_t)Cc * sizeof(float));
  _Float16* Wh   = (_Float16*)((char*)cbTg + (size_t)Cc * CBT_PITCH * sizeof(_Float16));
  float*    out  = (float*)d_out;

  rpq_cbsq_kernel<<<Cc / 256, 256, 0, stream>>>(cb, cbsq);
  rpq_prep_cb<<<(Qq * Cc) / 256, 256, 0, stream>>>(cb, cbTg);
  rpq_prep_w<<<(Qq * Dd) / 256, 256, 0, stream>>>(W, Wh);
  rpq_main_kernel<<<Mm / 32, 256, 0, stream>>>(X, Wh, cbTg, cbsq, out);
}
